// RefineMultiBoxLoss_24352464568756
// MI455X (gfx1250) — compile-verified, matrix operand order checked by reading, not run on hardware
//
#include <hip/hip_runtime.h>
#include <math.h>

#define B_ 32
#define P_ 16384
#define C_ 81
#define O_ 32
#define VAR0 0.1f
#define VAR1 0.2f
#define THRESH 0.5f

typedef __attribute__((ext_vector_type(16))) _Float16 v16h;
typedef __attribute__((ext_vector_type(8)))  float    v8f;

#if __has_builtin(__builtin_amdgcn_tensor_load_to_lds) && \
    __has_builtin(__builtin_amdgcn_s_wait_tensorcnt)
#define HAVE_TDM 1
typedef __attribute__((ext_vector_type(4))) unsigned int u32x4;
typedef __attribute__((ext_vector_type(4))) int          i32x4;
typedef __attribute__((ext_vector_type(8))) int          i32x8;

// Issue one TDM load of `nrows` rows x 81 f32 (contiguous, stride 81) from
// gsrc into LDS at ldsByteOff. D# per CDNA5 ISA 8.3/8.4 (2-D tile, type=2).
__device__ __forceinline__ void tdm_load_rows(const float* gsrc,
                                              unsigned ldsByteOff, int nrows) {
    unsigned long long ga = (unsigned long long)(uintptr_t)gsrc;
    u32x4 g0;
    g0[0] = 1u;                                  // count=1 (user descriptor)
    g0[1] = ldsByteOff;                          // lds_addr
    g0[2] = (unsigned)(ga & 0xFFFFFFFFu);        // global_addr[31:0]
    g0[3] = (unsigned)((ga >> 32) & 0x01FFFFFFu) // global_addr[56:32]
          | (2u << 30);                          // type=2 ("image")
    i32x8 g1;
    g1[0] = (int)(2u << 16);                     // data_size=2 -> 4 bytes
    g1[1] = (int)(((unsigned)C_ & 0xFFFFu) << 16);      // tensor_dim0[15:0]
    g1[2] = (int)(((unsigned)nrows & 0xFFFFu) << 16);   // dim0 hi=0 | tensor_dim1[15:0]
    g1[3] = (int)((unsigned)C_ << 16);           // dim1 hi=0 | tile_dim0=81
    g1[4] = (int)((unsigned)nrows & 0xFFFFu);    // tile_dim1=nrows, tile_dim2=0
    g1[5] = C_;                                  // tensor_dim0_stride lo32 = 81
    g1[6] = 0;                                   // stride0 hi | stride1 lo
    g1[7] = 0;
    i32x4 z4 = {0, 0, 0, 0};
#if defined(__clang_major__) && __clang_major__ >= 23
    i32x8 z8 = {0, 0, 0, 0, 0, 0, 0, 0};
    __builtin_amdgcn_tensor_load_to_lds(g0, g1, z4, z4, z8, 0);
#else
    __builtin_amdgcn_tensor_load_to_lds(g0, g1, z4, z4, 0);
#endif
}
#else
#define HAVE_TDM 0
#endif

// ---------------------------------------------------------------- init
__global__ void k_init(int* __restrict__ forced,
                       unsigned long long* __restrict__ objKey,
                       float* __restrict__ lossL,
                       int* __restrict__ numPos,
                       float* __restrict__ posCE) {
    size_t i = (size_t)blockIdx.x * blockDim.x + threadIdx.x;
    if (i < (size_t)B_ * P_) forced[i] = -1;
    if (i < B_ * O_)         objKey[i] = 0ull;
    if (i < B_)              { numPos[i] = 0; posCE[i] = 0.0f; }
    if (i == 0)              lossL[0] = 0.0f;
}

// ------------------------------------------------------------- matching
__global__ void __launch_bounds__(256)
k_match(const float* __restrict__ arm_loc, const float* __restrict__ priors,
        const float* __restrict__ tboxes,
        float* __restrict__ best_ov, int* __restrict__ best_idx,
        unsigned long long* __restrict__ objKey) {
    __shared__ float tb[O_][4];
    __shared__ float ta[O_];
    __shared__ unsigned long long skey[O_];
    const int blocksPerBatch = P_ / 256;
    const int b = blockIdx.x / blocksPerBatch;
    const int p = (blockIdx.x % blocksPerBatch) * 256 + threadIdx.x;
    const int t = threadIdx.x;
    if (t < O_ * 4) {
        int o = t >> 2, c = t & 3;
        tb[o][c] = tboxes[((size_t)b * O_ + o) * 4 + c];
    }
    if (t < O_) skey[t] = 0ull;
    __syncthreads();
    if (t < O_) ta[t] = (tb[t][2] - tb[t][0]) * (tb[t][3] - tb[t][1]);
    __syncthreads();

    const float* pr = priors + (size_t)p * 4;
    const float* al = arm_loc + ((size_t)b * P_ + p) * 4;
    float pw = pr[2], ph = pr[3];
    float cx = pr[0] + al[0] * VAR0 * pw;
    float cy = pr[1] + al[1] * VAR0 * ph;
    float w  = pw * __expf(al[2] * VAR1);
    float h  = ph * __expf(al[3] * VAR1);
    float x0 = cx - 0.5f * w, y0 = cy - 0.5f * h;
    float x1 = cx + 0.5f * w, y1 = cy + 0.5f * h;
    float areaB = w * h;

    float best = -1.0f; int bi = 0;
    #pragma unroll 4
    for (int o = 0; o < O_; ++o) {
        float ix = fminf(x1, tb[o][2]) - fmaxf(x0, tb[o][0]);
        float iy = fminf(y1, tb[o][3]) - fmaxf(y0, tb[o][1]);
        float inter = fmaxf(ix, 0.0f) * fmaxf(iy, 0.0f);
        float iou = inter / (ta[o] + areaB - inter);
        if (iou > best) { best = iou; bi = o; }   // first max wins (ascending o)
        unsigned long long key =
            ((unsigned long long)__float_as_uint(iou) << 24) |
            (unsigned long long)(0xFFFFFFu - (unsigned)p);
        atomicMax(&skey[o], key);
    }
    size_t gp = (size_t)b * P_ + p;
    best_ov[gp]  = best;
    best_idx[gp] = bi;
    __syncthreads();
    if (t < O_) atomicMax(&objKey[b * O_ + t], skey[t]);
}

// --------------------------------------------------------- forced match
__global__ void k_force(const unsigned long long* __restrict__ objKey,
                        float* __restrict__ best_ov, int* __restrict__ forced) {
    int i = blockIdx.x * blockDim.x + threadIdx.x;
    if (i >= B_ * O_) return;
    int b = i / O_, o = i % O_;
    unsigned long long key = objKey[i];
    int p = (int)(0xFFFFFFu - (unsigned)(key & 0xFFFFFFu));
    best_ov[(size_t)b * P_ + p] = 2.0f;
    atomicMax(&forced[(size_t)b * P_ + p], o);
}

// --------------------------- conf targets, encode, smooth-L1, pos count
__global__ void __launch_bounds__(256)
k_targets(const float* __restrict__ odm_loc, const float* __restrict__ arm_loc,
          const float* __restrict__ priors, const float* __restrict__ tboxes,
          const int* __restrict__ tlabels,
          const float* __restrict__ best_ov, const int* __restrict__ best_idx,
          const int* __restrict__ forced, int* __restrict__ conf_t,
          float* __restrict__ lossL, int* __restrict__ numPos) {
    __shared__ float sred[256];
    __shared__ int   scnt[256];
    const int blocksPerBatch = P_ / 256;
    const int b = blockIdx.x / blocksPerBatch;
    const int p = (blockIdx.x % blocksPerBatch) * 256 + threadIdx.x;
    size_t gp = (size_t)b * P_ + p;

    float ov = best_ov[gp];
    int f = forced[gp];
    int ti = (f >= 0) ? f : best_idx[gp];
    int cls = tlabels[b * O_ + ti];
    if (ov < THRESH) cls = 0;
    conf_t[gp] = cls;

    float l = 0.0f;
    if (cls > 0) {
        const float* pr = priors + (size_t)p * 4;
        const float* al = arm_loc + gp * 4;
        float pw = pr[2], ph = pr[3];
        float cx = pr[0] + al[0] * VAR0 * pw;
        float cy = pr[1] + al[1] * VAR0 * ph;
        float w  = pw * __expf(al[2] * VAR1);
        float h  = ph * __expf(al[3] * VAR1);
        const float* m = tboxes + ((size_t)b * O_ + ti) * 4;
        float tt[4];
        tt[0] = (0.5f * (m[0] + m[2]) - cx) / (VAR0 * w);
        tt[1] = (0.5f * (m[1] + m[3]) - cy) / (VAR0 * h);
        tt[2] = __logf((m[2] - m[0]) / w) / VAR1;
        tt[3] = __logf((m[3] - m[1]) / h) / VAR1;
        const float* ol = odm_loc + gp * 4;
        #pragma unroll
        for (int i = 0; i < 4; ++i) {
            float d = fabsf(ol[i] - tt[i]);
            l += (d < 1.0f) ? 0.5f * d * d : d - 0.5f;
        }
    }
    sred[threadIdx.x] = l;
    scnt[threadIdx.x] = (cls > 0) ? 1 : 0;
    __syncthreads();
    for (int s = 128; s > 0; s >>= 1) {
        if (threadIdx.x < s) {
            sred[threadIdx.x] += sred[threadIdx.x + s];
            scnt[threadIdx.x] += scnt[threadIdx.x + s];
        }
        __syncthreads();
    }
    if (threadIdx.x == 0) {
        atomicAdd(lossL, sred[0]);
        atomicAdd(&numPos[b], scnt[0]);
    }
}

// ------------------------- CE pass: TDM/LDS staging + WMMA logsumexp
// Block = 128 threads = 4 waves, 64 conf rows staged in LDS (TDM if present,
// else coalesced b128 copies). One wave handles 16 priors; lane pair (L,L+16)
// owns prior m = L&15 and supplies the 32 K-slots of A-row m; B = all ones so
// D[m,*] = rowsum(m) = sum(exp).
__global__ void __launch_bounds__(128)
k_ce(const float* __restrict__ odm_conf, const int* __restrict__ conf_t,
     unsigned* __restrict__ rankBits, float* __restrict__ posCE) {
    __shared__ float stage[64 * C_];             // 64 rows x 81 f32 = 20736 B
    const int wTile = threadIdx.x >> 5;
    const int lane  = threadIdx.x & 31;
    const int wave  = blockIdx.x * 4 + wTile;
    const int b     = wave / (P_ / 16);
    const int tile  = wave % (P_ / 16);
    const int m     = lane & 15;
    const int h     = lane >> 4;
    const int p     = tile * 16 + m;

    const float* gbase = odm_conf + (size_t)blockIdx.x * 64 * C_;
    __builtin_prefetch(gbase + 64 * C_, 0, 1);   // global_prefetch_b8 next tile

#if HAVE_TDM
    if (threadIdx.x < 32) {                      // wave 0 issues the block DMA
        tdm_load_rows(gbase, (unsigned)(uintptr_t)stage, 64);
        __builtin_amdgcn_s_wait_tensorcnt(0);
    }
    __syncthreads();
#else
    // fallback: fully coalesced b128 staging (tile base is 16B aligned)
    for (int i = threadIdx.x; i < (64 * C_) / 4; i += 128)
        ((float4*)stage)[i] = ((const float4*)gbase)[i];
    __syncthreads();
#endif

    const float* lrow = stage + (wTile * 16 + m) * C_;

    float v[48];
    float vmax = -INFINITY;
    #pragma unroll
    for (int c = 0; c < 3; ++c)
        #pragma unroll
        for (int i = 0; i < 16; ++i) {
            int cls = c * 32 + h * 16 + i;
            float x = (cls < C_) ? lrow[cls] : -INFINITY;
            v[c * 16 + i] = x;
            vmax = fmaxf(vmax, x);
        }
    vmax = fmaxf(vmax, __shfl_xor(vmax, 16));    // pair-wide max per prior

    v16h ones;
    #pragma unroll
    for (int i = 0; i < 16; ++i) ones[i] = (_Float16)1.0f;
    v8f acc = {};
    #pragma unroll
    for (int c = 0; c < 3; ++c) {
        v16h a;
        #pragma unroll
        for (int i = 0; i < 16; ++i) {
            float e = __expf(v[c * 16 + i] - vmax);  // pads -> exp(-inf) = 0
            a[i] = (_Float16)e;
        }
        acc = __builtin_amdgcn_wmma_f32_16x16x32_f16(
            false, a, false, ones, (short)0, acc, false, false);
    }

    // Extract rowsum(m) from the 16x16 f32 C/D layout.
    float a01 = (m & 1) ? acc[1] : acc[0];
    float a23 = (m & 1) ? acc[3] : acc[2];
    float a45 = (m & 1) ? acc[5] : acc[4];
    float a67 = (m & 1) ? acc[7] : acc[6];
    float a03 = (m & 2) ? a23 : a01;
    float a47 = (m & 2) ? a67 : a45;
    float vl  = (m & 4) ? a47 : a03;
    float vo  = __shfl_xor(vl, 16);
    float s   = (h == (m >> 3)) ? vl : vo;

    float lse  = vmax + __logf(s);
    int   cls  = conf_t[(size_t)b * P_ + p];
    float gold = lrow[cls];
    float ce   = lse - gold;
    bool  pos  = cls > 0;
    float rv   = pos ? 0.0f : ce;
    if (h == 0) rankBits[(size_t)b * P_ + p] = __float_as_uint(rv);

    __shared__ float sred[128];
    sred[threadIdx.x] = (pos && h == 0) ? ce : 0.0f;
    __syncthreads();
    for (int st = 64; st > 0; st >>= 1) {
        if (threadIdx.x < st) sred[threadIdx.x] += sred[threadIdx.x + st];
        __syncthreads();
    }
    if (threadIdx.x == 0) atomicAdd(&posCE[b], sred[0]);
}

// --------------------- hard-negative mining: radix-select top-k per batch
__global__ void __launch_bounds__(256)
k_select(const unsigned* __restrict__ rankBits, const int* __restrict__ numPos,
         const float* __restrict__ posCE, float* __restrict__ lossC) {
    const int b = blockIdx.x;
    const int t = threadIdx.x;
    __shared__ unsigned hist[256];
    __shared__ unsigned sh_t;
    __shared__ int      sh_r;

    int k = 3 * numPos[b];
    if (k > P_ - 1) k = P_ - 1;
    if (k <= 0) { if (t == 0) lossC[b] = posCE[b]; return; }
    if (t == 0) { sh_t = 0u; sh_r = k; }
    __syncthreads();

    const unsigned* keys = rankBits + (size_t)b * P_;
    for (int pass = 3; pass >= 0; --pass) {
        hist[t] = 0u;
        __syncthreads();
        unsigned tcur = sh_t;
        int hs = (pass + 1) * 8;
        for (int i = t; i < P_; i += 256) {
            unsigned key = keys[i];
            bool match = (pass == 3) ? true : ((key >> hs) == (tcur >> hs));
            if (match) atomicAdd(&hist[(key >> (pass * 8)) & 255u], 1u);
        }
        __syncthreads();
        if (t == 0) {
            int r = sh_r;
            unsigned tt = sh_t;
            for (int bin = 255; bin >= 0; --bin) {
                int c = (int)hist[bin];
                if (r > c) r -= c;
                else { tt |= ((unsigned)bin) << (pass * 8); break; }
            }
            sh_r = r; sh_t = tt;
        }
        __syncthreads();
    }
    unsigned tfin = sh_t;
    int r = sh_r;   // ties at threshold share one value -> r * val(t) exact
    float sum = 0.0f;
    for (int i = t; i < P_; i += 256) {
        unsigned key = keys[i];
        if (key > tfin) sum += __uint_as_float(key);
    }
    __shared__ float sred[256];
    sred[t] = sum;
    __syncthreads();
    for (int st = 128; st > 0; st >>= 1) {
        if (t < st) sred[t] += sred[t + st];
        __syncthreads();
    }
    if (t == 0)
        lossC[b] = posCE[b] + sred[0] + (float)r * __uint_as_float(tfin);
}

// ---------------------------------------------------------------- final
__global__ void k_final(const int* __restrict__ numPos,
                        const float* __restrict__ lossL,
                        const float* __restrict__ lossC,
                        float* __restrict__ out) {
    if (threadIdx.x == 0 && blockIdx.x == 0) {
        int n = 0; float lc = 0.0f;
        for (int b = 0; b < B_; ++b) { n += numPos[b]; lc += lossC[b]; }
        float N = (float)n;
        out[0] = lossL[0] / N;
        out[1] = lc / N;
    }
}

// ------------------------------------------------------------- launcher
extern "C" void kernel_launch(void* const* d_in, const int* in_sizes, int n_in,
                              void* d_out, int out_size, void* d_ws, size_t ws_size,
                              hipStream_t stream) {
    const float* odm_loc  = (const float*)d_in[0];
    const float* odm_conf = (const float*)d_in[1];
    const float* arm_loc  = (const float*)d_in[2];
    const float* priors   = (const float*)d_in[3];
    const float* tboxes   = (const float*)d_in[4];
    const int*   tlabels  = (const int*)d_in[5];
    float* out = (float*)d_out;

    char* ws = (char*)d_ws;
    auto carve = [&](size_t bytes) -> char* {
        char* r = ws; ws += (bytes + 255) & ~(size_t)255; return r;
    };
    float*    best_ov  = (float*)carve(sizeof(float) * B_ * P_);
    int*      best_idx = (int*)  carve(sizeof(int)   * B_ * P_);
    int*      forced   = (int*)  carve(sizeof(int)   * B_ * P_);
    int*      conf_t   = (int*)  carve(sizeof(int)   * B_ * P_);
    unsigned* rankBits = (unsigned*)carve(sizeof(unsigned) * B_ * P_);
    unsigned long long* objKey = (unsigned long long*)carve(8ull * B_ * O_);
    int*   numPos = (int*)  carve(sizeof(int)   * B_);
    float* posCE  = (float*)carve(sizeof(float) * B_);
    float* lossC  = (float*)carve(sizeof(float) * B_);
    float* lossL  = (float*)carve(sizeof(float) * 4);

    const int bpP = (B_ * P_) / 256;
    k_init<<<bpP, 256, 0, stream>>>(forced, objKey, lossL, numPos, posCE);
    k_match<<<bpP, 256, 0, stream>>>(arm_loc, priors, tboxes,
                                     best_ov, best_idx, objKey);
    k_force<<<(B_ * O_ + 255) / 256, 256, 0, stream>>>(objKey, best_ov, forced);
    k_targets<<<bpP, 256, 0, stream>>>(odm_loc, arm_loc, priors, tboxes, tlabels,
                                       best_ov, best_idx, forced, conf_t,
                                       lossL, numPos);
    k_ce<<<(B_ * P_) / 64, 128, 0, stream>>>(odm_conf, conf_t, rankBits, posCE);
    k_select<<<B_, 256, 0, stream>>>(rankBits, numPos, posCE, lossC);
    k_final<<<1, 32, 0, stream>>>(numPos, lossL, lossC, out);
}
